// RNN_85469849191027
// MI455X (gfx1250) — matrix operand order repaired and checked by reading, not compiled, not run on hardware
//
#include <hip/hip_runtime.h>

// ---------------- problem constants ----------------
#define B_   8192
#define T_   10
#define F_   561
#define FP_  576          // F padded to multiple of 32
#define H_   256
#define O_   12
#define G4_  1024         // 4*H
#define BT_  (B_ * T_)

typedef _Float16 half8  __attribute__((ext_vector_type(8)));
typedef _Float16 half16 __attribute__((ext_vector_type(16)));
typedef float    floatx8 __attribute__((ext_vector_type(8)));

// ---- WMMA fragment loaders (16x16x32 f16; contiguous k order, same for A and B) ----
// A-matrix 16x32: lane l (=L%16) holds row M=l; hi=L/16 selects K-halves:
// elems j -> K = hi*16 + j (contiguous)
__device__ __forceinline__ half16 frag_a(const _Float16* p /* row base + hi*16 + kk */) {
    return *(const half16*)(p);
}
// B-matrix 32x16: lane l holds column N=l; elems j -> K = hi*16 + j (contiguous)
__device__ __forceinline__ half16 frag_b(const _Float16* p /* row base + hi*16 + kk */) {
    return *(const half16*)(p);
}

__device__ __forceinline__ floatx8 wmma_f16(half16 a, half16 b, floatx8 c) {
    return __builtin_amdgcn_wmma_f32_16x16x32_f16(false, a, false, b, (short)0, c, false, false);
}

__device__ __forceinline__ float sigm_(float x) { return 1.0f / (1.0f + __expf(-x)); }

// ---------------- conversion / prep kernels ----------------
__global__ void k_pad_cvt(const float* __restrict__ src, _Float16* __restrict__ dst,
                          int rows, int scols, int dcols) {
    long long i = (long long)blockIdx.x * blockDim.x + threadIdx.x;
    long long tot = (long long)rows * dcols;
    if (i >= tot) return;
    int r = (int)(i / dcols), c = (int)(i % dcols);
    dst[i] = (c < scols) ? (_Float16)src[(long long)r * scols + c] : (_Float16)0.0f;
}

__global__ void k_cvt(const float* __restrict__ src, _Float16* __restrict__ dst, int n) {
    int i = blockIdx.x * blockDim.x + threadIdx.x;
    if (i < n) dst[i] = (_Float16)src[i];
}

// dst[n][0:cols]=s0[n], dst[n][cols:2cols]=s1[n]
__global__ void k_concat2(const float* __restrict__ s0, const float* __restrict__ s1,
                          _Float16* __restrict__ dst, int rows, int cols) {
    int i = blockIdx.x * blockDim.x + threadIdx.x;
    int tot = rows * 2 * cols;
    if (i >= tot) return;
    int r = i / (2 * cols), c = i % (2 * cols);
    float v = (c < cols) ? s0[r * cols + c] : s1[r * cols + (c - cols)];
    dst[i] = (_Float16)v;
}

__global__ void k_vadd(const float* __restrict__ a, const float* __restrict__ b,
                       float* __restrict__ o, int n) {
    int i = blockIdx.x * blockDim.x + threadIdx.x;
    if (i < n) o[i] = a[i] + b[i];
}

// ---------------- generic f16 GEMM: C[M,N] = A[M,K] * W[N,K]^T (+bias) ----------------
// one wave -> 16x64 tile (1 A-frag, 4 B-frags, 4 WMMA per K-step)
__global__ void k_gemm_f16(const _Float16* __restrict__ A,
                           const _Float16* __restrict__ W, int K,
                           const float* __restrict__ bias,
                           _Float16* __restrict__ C, int N, int M) {
    int wave = (blockIdx.x * blockDim.x + threadIdx.x) >> 5;
    int lane = threadIdx.x & 31;
    int nblk = N >> 6;
    int m_idx = wave / nblk;
    int n_blk = wave % nblk;
    if (m_idx * 16 >= M) return;                 // wave-uniform
    int l = lane & 15, hi = lane >> 4;
    int m0 = m_idx * 16, n0 = n_blk * 64;

    floatx8 acc[4] = {};
    const _Float16* arow = A + (size_t)(m0 + l) * K + hi * 16;
    const _Float16* wrow0 = W + (size_t)(n0 +  0 + l) * K + hi * 16;
    const _Float16* wrow1 = W + (size_t)(n0 + 16 + l) * K + hi * 16;
    const _Float16* wrow2 = W + (size_t)(n0 + 32 + l) * K + hi * 16;
    const _Float16* wrow3 = W + (size_t)(n0 + 48 + l) * K + hi * 16;

    for (int kk = 0; kk < K; kk += 32) {
        half16 a = frag_a(arow + kk);
        acc[0] = wmma_f16(a, frag_b(wrow0 + kk), acc[0]);
        acc[1] = wmma_f16(a, frag_b(wrow1 + kk), acc[1]);
        acc[2] = wmma_f16(a, frag_b(wrow2 + kk), acc[2]);
        acc[3] = wmma_f16(a, frag_b(wrow3 + kk), acc[3]);
    }
#pragma unroll
    for (int j = 0; j < 4; j++) {
        int n = n0 + j * 16 + l;
        float bv = bias ? bias[n] : 0.0f;
#pragma unroll
        for (int r = 0; r < 8; r++) {
            int m = m0 + r + 8 * hi;
            C[(size_t)m * N + n] = (_Float16)(acc[j][r] + bv);
        }
    }
}

// ---------------- fused recurrent-GEMM + LSTM cell ----------------
// gates[B,4H] = [A0|A1][B,K] * W[4H,K]^T + Gpre + bias ; then cell update.
// One wave -> 16 batch rows x 16 h-cols, all 4 gate tiles in registers.
__global__ void k_lstm_cell(const _Float16* __restrict__ A0,
                            const _Float16* __restrict__ A1, int K0, int K,
                            const _Float16* __restrict__ W,          // [1024,K]
                            const _Float16* __restrict__ Gpre, int ldg,
                            const float* __restrict__ bias,          // [1024] or null
                            float* __restrict__ Cst,                 // c state [B,H] fp32
                            float* __restrict__ Hf32,                // h out fp32 [B,H]
                            _Float16* __restrict__ Hf16,             // h out f16 (next-step GEMM)
                            _Float16* __restrict__ Hf16b,            // optional 2nd f16 copy (Hs)
                            int M) {
    int wave = (blockIdx.x * blockDim.x + threadIdx.x) >> 5;
    int lane = threadIdx.x & 31;
    int m_idx = wave >> 4;              // H/16 == 16 n-tiles
    int n_idx = wave & 15;
    if (m_idx * 16 >= M) return;
    int l = lane & 15, hi = lane >> 4;
    int m0 = m_idx * 16, n0 = n_idx * 16;

    floatx8 acc[4] = {};                // i, f, g, o
    const _Float16* a0row = A0 + (size_t)(m0 + l) * H_ + hi * 16;
    const _Float16* a1row = A1 ? (A1 + (size_t)(m0 + l) * H_ + hi * 16) : (const _Float16*)0;
    const _Float16* wrow[4];
#pragma unroll
    for (int q = 0; q < 4; q++)
        wrow[q] = W + (size_t)(q * H_ + n0 + l) * K + hi * 16;

    for (int kk = 0; kk < K; kk += 32) {
        const _Float16* ap = (kk < K0) ? (a0row + kk) : (a1row + (kk - K0));
        half16 a = frag_a(ap);
#pragma unroll
        for (int q = 0; q < 4; q++)
            acc[q] = wmma_f16(a, frag_b(wrow[q] + kk), acc[q]);
    }

    int h = n0 + l;
    float b0 = 0.f, b1 = 0.f, b2 = 0.f, b3 = 0.f;
    if (bias) { b0 = bias[h]; b1 = bias[H_ + h]; b2 = bias[2 * H_ + h]; b3 = bias[3 * H_ + h]; }
#pragma unroll
    for (int r = 0; r < 8; r++) {
        int m = m0 + r + 8 * hi;
        float gi = acc[0][r] + b0, gf = acc[1][r] + b1;
        float gg = acc[2][r] + b2, go = acc[3][r] + b3;
        if (Gpre) {
            const _Float16* gp = Gpre + (size_t)m * ldg;
            gi += (float)gp[h];          gf += (float)gp[H_ + h];
            gg += (float)gp[2 * H_ + h]; go += (float)gp[3 * H_ + h];
        }
        float iv = sigm_(gi), fv = sigm_(gf), gv = tanhf(gg), ov = sigm_(go);
        size_t idx = (size_t)m * H_ + h;
        float cn = fv * Cst[idx] + iv * gv;
        Cst[idx] = cn;
        float hv = ov * tanhf(cn);
        Hf32[idx] = hv;
        _Float16 hh = (_Float16)hv;
        Hf16[idx] = hh;
        if (Hf16b) Hf16b[idx] = hh;
    }
}

// ---------------- tiny output projection: out[b,t,o] = Hs[t,b,:]·Wout[o,:] + bout ----------------
__global__ void k_out_proj(const _Float16* __restrict__ Hs, const float* __restrict__ Wout,
                           const float* __restrict__ bout, float* __restrict__ out, int tot) {
    int i = blockIdx.x * blockDim.x + threadIdx.x;
    if (i >= tot) return;
    int o = i % O_;
    int bt = i / O_;
    int t = bt % T_, b = bt / T_;
    const _Float16* hrow = Hs + ((size_t)t * B_ + b) * H_;
    const float* wrow = Wout + (size_t)o * H_;
    float s = 0.0f;
#pragma unroll 8
    for (int h = 0; h < H_; h++) s += (float)hrow[h] * wrow[h];
    out[i] = s + bout[o];
}

// ---------------- host orchestration ----------------
extern "C" void kernel_launch(void* const* d_in, const int* in_sizes, int n_in,
                              void* d_out, int out_size, void* d_ws, size_t ws_size,
                              hipStream_t stream) {
    const float* inputs = (const float*)d_in[0];
    const float* W_inp  = (const float*)d_in[1];
    const float* b_inp  = (const float*)d_in[2];
    const float* Wih0   = (const float*)d_in[3];
    const float* Whh0   = (const float*)d_in[4];
    const float* bih0   = (const float*)d_in[5];
    const float* bhh0   = (const float*)d_in[6];
    const float* Wih1   = (const float*)d_in[7];
    const float* Whh1   = (const float*)d_in[8];
    const float* bih1   = (const float*)d_in[9];
    const float* bhh1   = (const float*)d_in[10];
    const float* W_out  = (const float*)d_in[11];
    const float* b_out  = (const float*)d_in[12];
    float* out = (float*)d_out;

    // ---- workspace carve-up ----
    char* ws = (char*)d_ws;
    size_t off = 0;
    auto take = [&](size_t bytes) { char* p = ws + off; off += (bytes + 255) & ~(size_t)255; return p; };
    _Float16* Xin    = (_Float16*)take((size_t)BT_ * FP_ * 2);   // padded f16 inputs
    _Float16* Winp   = (_Float16*)take((size_t)H_ * FP_ * 2);
    _Float16* Xp     = (_Float16*)take((size_t)BT_ * H_ * 2);    // input projection (f16)
    _Float16* G0     = (_Float16*)take((size_t)BT_ * G4_ * 2);   // precomputed layer0 gate input
    _Float16* Wih0h  = (_Float16*)take((size_t)G4_ * H_ * 2);
    _Float16* Whh0h  = (_Float16*)take((size_t)G4_ * H_ * 2);
    _Float16* Wcat1h = (_Float16*)take((size_t)G4_ * 2 * H_ * 2); // [Wih1|Whh1]
    float*    bias0  = (float*)take((size_t)G4_ * 4);
    float*    bias1  = (float*)take((size_t)G4_ * 4);
    _Float16* h0b[2] = { (_Float16*)take((size_t)B_ * H_ * 2), (_Float16*)take((size_t)B_ * H_ * 2) };
    _Float16* h1b[2] = { (_Float16*)take((size_t)B_ * H_ * 2), (_Float16*)take((size_t)B_ * H_ * 2) };
    _Float16* Hs     = (_Float16*)take((size_t)T_ * B_ * H_ * 2);

    // ---- d_out layout: outputs[B,T,O] | h_n[2,B,H] | c_n[2,B,H] ----
    float* hn0 = out + (size_t)B_ * T_ * O_;
    float* hn1 = hn0 + (size_t)B_ * H_;
    float* cn0 = hn1 + (size_t)B_ * H_;
    float* cn1 = cn0 + (size_t)B_ * H_;

    // zero initial states (h f16 ping buffers + c state living in d_out c_n region)
    hipMemsetAsync(h0b[0], 0, (size_t)B_ * H_ * 2, stream);
    hipMemsetAsync(h1b[0], 0, (size_t)B_ * H_ * 2, stream);
    hipMemsetAsync(cn0, 0, (size_t)2 * B_ * H_ * 4, stream);

    // ---- weight / input conversion ----
    {
        long long tot = (long long)BT_ * FP_;
        k_pad_cvt<<<(int)((tot + 255) / 256), 256, 0, stream>>>(inputs, Xin, BT_, F_, FP_);
        int tw = H_ * FP_;
        k_pad_cvt<<<(tw + 255) / 256, 256, 0, stream>>>(W_inp, Winp, H_, F_, FP_);
        int t0 = G4_ * H_;
        k_cvt<<<(t0 + 255) / 256, 256, 0, stream>>>(Wih0, Wih0h, t0);
        k_cvt<<<(t0 + 255) / 256, 256, 0, stream>>>(Whh0, Whh0h, t0);
        int tc = G4_ * 2 * H_;
        k_concat2<<<(tc + 255) / 256, 256, 0, stream>>>(Wih1, Whh1, Wcat1h, G4_, H_);
        k_vadd<<<(G4_ + 255) / 256, 256, 0, stream>>>(bih0, bhh0, bias0, G4_);
        k_vadd<<<(G4_ + 255) / 256, 256, 0, stream>>>(bih1, bhh1, bias1, G4_);
    }

    // ---- big parallel GEMMs (WMMA f16) ----
    // Xp = Xin[81920,576] * Winp[256,576]^T + b_inp
    {
        int waves = (BT_ / 16) * (H_ / 64);
        k_gemm_f16<<<waves / 8, 256, 0, stream>>>(Xin, Winp, FP_, b_inp, Xp, H_, BT_);
    }
    // G0 = Xp[81920,256] * Wih0[1024,256]^T + bias0
    {
        int waves = (BT_ / 16) * (G4_ / 64);
        k_gemm_f16<<<waves / 8, 256, 0, stream>>>(Xp, Wih0h, H_, bias0, G0, G4_, BT_);
    }

    // ---- recurrence: 10 steps x 2 fused GEMM+cell kernels ----
    int cellWaves = (B_ / 16) * (H_ / 16);   // 8192 waves -> 1024 blocks
    for (int t = 0; t < T_; t++) {
        _Float16* h0p = h0b[t & 1];       _Float16* h0c = h0b[(t + 1) & 1];
        _Float16* h1p = h1b[t & 1];       _Float16* h1c = h1b[(t + 1) & 1];
        // layer 0: gates = h0p*Whh0^T + G0[:,t,:]   (bias already folded into G0)
        k_lstm_cell<<<cellWaves / 8, 256, 0, stream>>>(
            h0p, (const _Float16*)0, H_, H_, Whh0h,
            G0 + (size_t)t * G4_, T_ * G4_, (const float*)0,
            cn0, hn0, h0c, (_Float16*)0, B_);
        // layer 1: gates = [h0c|h1p]*[Wih1|Whh1]^T + bias1
        k_lstm_cell<<<cellWaves / 8, 256, 0, stream>>>(
            h0c, h1p, H_, 2 * H_, Wcat1h,
            (const _Float16*)0, 0, bias1,
            cn1, hn1, h1c, Hs + (size_t)t * B_ * H_, B_);
    }

    // ---- final tiny projection ----
    int totO = B_ * T_ * O_;
    k_out_proj<<<(totO + 255) / 256, 256, 0, stream>>>(Hs, W_out, b_out, out, totO);
}
